// TriangleAttentionOpt_2869038153748
// MI455X (gfx1250) — matrix operand order repaired and checked by reading, compiled, not run on hardware
//
#include <hip/hip_runtime.h>
#include <hip/hip_bf16.h>
#include <math.h>

// ---------------------------------------------------------------------------
// Triangle attention (per_row) for MI455X / gfx1250, wave32 + WMMA f16.
//   N=384 residues, C=128 pair channels, H=4 heads, D=32 key dim.
// Pipeline: pack_w -> ln -> kv_proj -> att   (all on `stream`).
// ---------------------------------------------------------------------------

#define NRES 384
#define CDIM 128
#define NHEAD 4
#define KDIM 32
#define NN (NRES * NRES)

typedef _Float16 f16_t;
typedef __attribute__((ext_vector_type(16))) _Float16 v16h;
typedef __attribute__((ext_vector_type(8)))  _Float16 h8;
typedef __attribute__((ext_vector_type(8)))  float    v8f;

// D = A(16x32 f16) * B(32x16 f16) + C(16x16 f32)
__device__ __forceinline__ v8f wmma16(v16h a, v16h b, v8f c) {
  return __builtin_amdgcn_wmma_f32_16x16x32_f16(false, a, false, b,
                                                (short)0, c, false, false);
}

// A-fragment (16x32 f16) load. Caller passes p = base + (lane&15)*ld + (lane>>4)*16
// (+ k-step offset). halves[0..15] = K kbase..kbase+15 (contiguous)
__device__ __forceinline__ v16h load_a16x32(const f16_t* p) {
  union { v16h v; h8 h[2]; } u;
  u.h[0] = *(const h8*)(p);
  u.h[1] = *(const h8*)(p + 8);
  return u.v;
}

// ---------------------------------------------------------------------------
// Kernel 1: pack weight matrices (K-major [128][128] f32) into B-fragment
// f16 layout: WP[(ks*8+nt)*32 + lane][i] = W[ks*32 + kbase + i][nt*16 + col]
// with col = lane&15, kbase = (lane>>4)*16.  q_w gets 1/sqrt(32) folded in.
// ---------------------------------------------------------------------------
__global__ __launch_bounds__(1024) void pack_w_kernel(
    const float* __restrict__ qw, const float* __restrict__ kw,
    const float* __restrict__ vw, const float* __restrict__ gw,
    const float* __restrict__ ow, f16_t* __restrict__ WP) {
  int which = blockIdx.x;
  const float* W;
  float scale = 1.0f;
  if (which == 0)      { W = qw; scale = 0.17677669529663687f; }
  else if (which == 1) { W = kw; }
  else if (which == 2) { W = vw; }
  else if (which == 3) { W = gw; }
  else                 { W = ow; }
  int t = threadIdx.x;
  int lane = t & 31, nt = (t >> 5) & 7, ks = t >> 8;
  int col = lane & 15, kb = (lane >> 4) * 16;
  f16_t* dst = WP + (size_t)which * (CDIM * CDIM) +
               (size_t)(((ks * 8 + nt) * 32 + lane) * 16);
#pragma unroll
  for (int i = 0; i < 16; ++i)
    dst[i] = (f16_t)(W[(ks * 32 + kb + i) * CDIM + nt * 16 + col] * scale);
}

// ---------------------------------------------------------------------------
// Kernel 2: LayerNorm -> x (f16) + nb_bias[h][q][k] (f32).
// One wave per (i,j) position; 8 positions per 256-thread block.
// ---------------------------------------------------------------------------
__global__ __launch_bounds__(256) void ln_kernel(
    const float* __restrict__ pair_act, const float* __restrict__ ln_scale,
    const float* __restrict__ ln_bias, const float* __restrict__ f2w,
    f16_t* __restrict__ X16, float* __restrict__ NB) {
  int tid = threadIdx.x, lane = tid & 31, w = tid >> 5;
  size_t p = (size_t)blockIdx.x * 8 + w;          // p = i*NRES + j
  const float4* src = (const float4*)(pair_act + p * CDIM);
  float4 x = src[lane];
  float s  = x.x + x.y + x.z + x.w;
  float sq = x.x * x.x + x.y * x.y + x.z * x.z + x.w * x.w;
#pragma unroll
  for (int off = 16; off >= 1; off >>= 1) {
    s  += __shfl_xor(s, off, 32);
    sq += __shfl_xor(sq, off, 32);
  }
  float mu  = s * (1.0f / CDIM);
  float var = sq * (1.0f / CDIM) - mu * mu;
  float rs  = rsqrtf(var + 1e-5f);
  int c = lane * 4;
  float y0 = (x.x - mu) * rs * ln_scale[c + 0] + ln_bias[c + 0];
  float y1 = (x.y - mu) * rs * ln_scale[c + 1] + ln_bias[c + 1];
  float y2 = (x.z - mu) * rs * ln_scale[c + 2] + ln_bias[c + 2];
  float y3 = (x.w - mu) * rs * ln_scale[c + 3] + ln_bias[c + 3];
  union { struct { f16_t a, b, c, d; } e; unsigned long long u; } pk;
  pk.e.a = (f16_t)y0; pk.e.b = (f16_t)y1; pk.e.c = (f16_t)y2; pk.e.d = (f16_t)y3;
  *(unsigned long long*)(X16 + p * CDIM + c) = pk.u;
  // nb_bias: 4 head dots over the 128 channels
  float acc[4];
#pragma unroll
  for (int h = 0; h < 4; ++h)
    acc[h] = y0 * f2w[(c + 0) * 4 + h] + y1 * f2w[(c + 1) * 4 + h] +
             y2 * f2w[(c + 2) * 4 + h] + y3 * f2w[(c + 3) * 4 + h];
#pragma unroll
  for (int h = 0; h < 4; ++h)
#pragma unroll
    for (int off = 16; off >= 1; off >>= 1)
      acc[h] += __shfl_xor(acc[h], off, 32);
  if (lane == 0) {
#pragma unroll
    for (int h = 0; h < 4; ++h) NB[(size_t)h * NN + p] = acc[h];
  }
}

// ---------------------------------------------------------------------------
// Kernel 3: per-row K/V projection (WMMA) + repack into WMMA B-fragments.
//   KP per (b,h): 24 k-tiles of B[c][kpos]  (for S = Q @ K^T)
//   VP per (b,h): 12 k-blocks x 2 c-tiles of B[kpos][c]  (for O = P @ V)
// Dynamic LDS: X_b (96 KB) + staging (96 KB) = 192 KB of the 320 KB WGP LDS.
// Loop order nt -> mt so each B-fragment global load feeds 3 WMMAs; the 12
// A-fragments (per wave) are hoisted across both the K and V passes.
// ---------------------------------------------------------------------------
__global__ __launch_bounds__(256) void kv_proj_kernel(
    const f16_t* __restrict__ X16, const f16_t* __restrict__ WPk,
    const f16_t* __restrict__ WPv, f16_t* __restrict__ KP,
    f16_t* __restrict__ VP) {
  extern __shared__ f16_t sm[];
  f16_t* Xs = sm;                  // [384][128]
  f16_t* Ss = sm + NRES * CDIM;    // [384][128] projection staging
  int b = blockIdx.x, tid = threadIdx.x;
  int lane = tid & 31, w = tid >> 5;

  const h8* src = (const h8*)(X16 + (size_t)b * NRES * CDIM);
  h8* dst = (h8*)Xs;
  for (int i = tid; i < NRES * CDIM / 8; i += 256) dst[i] = src[i];
  __syncthreads();

  // hoist all A-fragments for this wave's 3 M-tiles (reused in both passes)
  v16h a[3][4];
#pragma unroll
  for (int m = 0; m < 3; ++m) {
    int mt = w * 3 + m;
#pragma unroll
    for (int ks = 0; ks < 4; ++ks)
      a[m][ks] = load_a16x32(Xs + (mt * 16 + (lane & 15)) * CDIM + ks * 32 +
                             (lane >> 4) * 16);
  }

  for (int pass = 0; pass < 2; ++pass) {
    const f16_t* WP = pass ? WPv : WPk;
#pragma unroll
    for (int nt = 0; nt < 8; ++nt) {
      // load the 4 B-fragments of this output tile once...
      v16h bf[4];
#pragma unroll
      for (int ks = 0; ks < 4; ++ks)
        bf[ks] = *(const v16h*)(WP + (size_t)(((ks * 8 + nt) * 32 + lane) * 16));
      // ...and reuse them across the wave's 3 M-tiles
#pragma unroll
      for (int m = 0; m < 3; ++m) {
        int mt = w * 3 + m;
        v8f acc = {};
#pragma unroll
        for (int ks = 0; ks < 4; ++ks) acc = wmma16(a[m][ks], bf[ks], acc);
        int mrow = mt * 16 + (lane >> 4) * 8;
        int ncol = nt * 16 + (lane & 15);
#pragma unroll
        for (int v = 0; v < 8; ++v)
          Ss[(mrow + v) * CDIM + ncol] = (f16_t)acc[v];
      }
    }
    __syncthreads();
    if (pass == 0) {
      // K^T fragments: B[c][kpos] -> contiguous 16 halfs per lane in Ss
      for (int idx = tid; idx < NHEAD * 24 * 32; idx += 256) {
        int lp = idx & 31, kt = (idx >> 5) % 24, h = idx / (32 * 24);
        int col = lp & 15, kb = (lp >> 4) * 16;
        const h8* sp = (const h8*)(Ss + (kt * 16 + col) * CDIM + h * 32 + kb);
        h8* dp = (h8*)(KP + (size_t)((((size_t)b * NHEAD + h) * 24 + kt) * 32 + lp) * 16);
        dp[0] = sp[0]; dp[1] = sp[1];
      }
    } else {
      // V fragments: B[kpos][c] -> strided gather from Ss
      for (int idx = tid; idx < NHEAD * 12 * 2 * 32; idx += 256) {
        int lp = idx & 31, ct = (idx >> 5) & 1, kblk = (idx >> 6) % 12,
            h = idx / (64 * 12);
        int col = lp & 15, kb = (lp >> 4) * 16;
        f16_t tmp[16];
#pragma unroll
        for (int i = 0; i < 16; ++i)
          tmp[i] = Ss[(kblk * 32 + kb + i) * CDIM + h * 32 + ct * 16 + col];
        h8* dp = (h8*)(VP + (size_t)(((((size_t)b * NHEAD + h) * 12 + kblk) * 2 + ct) * 32 + lp) * 16);
        dp[0] = *(h8*)&tmp[0]; dp[1] = *(h8*)&tmp[8];
      }
    }
    __syncthreads();
  }
}

// ---------------------------------------------------------------------------
// Kernel 4: fused flash attention per (b, 16-row q-tile); 4 waves = 4 heads.
// KP/VP fragments are double-buffered in registers: the kb+1 loads are issued
// before the kb WMMAs + softmax, hiding global latency behind a full iteration.
// ---------------------------------------------------------------------------
__global__ __launch_bounds__(128) void att_kernel(
    const f16_t* __restrict__ X16, const float* __restrict__ NB,
    const float* __restrict__ mask, const f16_t* __restrict__ KP,
    const f16_t* __restrict__ VP, const f16_t* __restrict__ WPq,
    const f16_t* __restrict__ WPg, const f16_t* __restrict__ WPo,
    const float* __restrict__ gate_b, const float* __restrict__ o_b,
    float* __restrict__ out) {
  __shared__ f16_t Xs[16 * CDIM];
  __shared__ f16_t Qs[NHEAD][16 * KDIM];
  __shared__ f16_t Ps[NHEAD][16 * KDIM];
  __shared__ float outAcc[16 * CDIM];

  int b = blockIdx.y, q0 = blockIdx.x * 16;
  int tid = threadIdx.x, lane = tid & 31, h = tid >> 5;
  int col = lane & 15, hi = lane >> 4;

  const h8* xsrc = (const h8*)(X16 + ((size_t)b * NRES + q0) * CDIM);
  h8* xdst = (h8*)Xs;
  for (int i = tid; i < 16 * CDIM / 8; i += 128) xdst[i] = xsrc[i];
  for (int i = tid; i < 16 * CDIM; i += 128) outAcc[i] = o_b[i & (CDIM - 1)];
  __syncthreads();

  // A-fragments of the X tile, one per 32-wide K step
  v16h xa[4];
#pragma unroll
  for (int ks = 0; ks < 4; ++ks)
    xa[ks] = load_a16x32(Xs + (lane & 15) * CDIM + ks * 32 + (lane >> 4) * 16);

  // Q projection (scale folded into WPq); bounce C-frag -> A-frag via LDS
#pragma unroll
  for (int nt2 = 0; nt2 < 2; ++nt2) {
    int nt = h * 2 + nt2;
    v8f acc = {};
#pragma unroll
    for (int ks = 0; ks < 4; ++ks) {
      v16h bf = *(const v16h*)(WPq + (size_t)(((ks * 8 + nt) * 32 + lane) * 16));
      acc = wmma16(xa[ks], bf, acc);
    }
#pragma unroll
    for (int v = 0; v < 8; ++v)
      Qs[h][(v + 8 * hi) * KDIM + nt2 * 16 + col] = (f16_t)acc[v];
  }
  // Gate projection: C-frag layout matches the O accumulator -> stays in regs
  v8f gf[2];
#pragma unroll
  for (int nt2 = 0; nt2 < 2; ++nt2) {
    int nt = h * 2 + nt2;
    v8f acc = {};
#pragma unroll
    for (int ks = 0; ks < 4; ++ks) {
      v16h bf = *(const v16h*)(WPg + (size_t)(((ks * 8 + nt) * 32 + lane) * 16));
      acc = wmma16(xa[ks], bf, acc);
    }
    gf[nt2] = acc;
  }
  __asm__ volatile("s_wait_dscnt 0x0" ::: "memory");
  v16h qa = load_a16x32(Qs[h] + (lane & 15) * KDIM + (lane >> 4) * 16);

  // flash loop over 12 k-blocks of 32 keys, register double-buffered
  float rmax[8], lsum[8];
#pragma unroll
  for (int v = 0; v < 8; ++v) { rmax[v] = -3.0e38f; lsum[v] = 0.0f; }
  v8f oc0 = {}, oc1 = {};
  const size_t nbBase = (size_t)h * NN;
  const v16h* kpB = (const v16h*)(KP + (size_t)((size_t)b * NHEAD + h) * 24 * 32 * 16);
  const v16h* vpB = (const v16h*)(VP + (size_t)((size_t)b * NHEAD + h) * 12 * 2 * 32 * 16);

  v16h kpc0 = kpB[lane];
  v16h kpc1 = kpB[32 + lane];
  v16h vpc0 = vpB[lane];
  v16h vpc1 = vpB[32 + lane];

  for (int kb = 0; kb < 12; ++kb) {
    // prefetch next k-block's fragments (clamped, branch-free)
    int nk = kb < 11 ? kb + 1 : 11;
    v16h kpn0 = kpB[(nk * 2 + 0) * 32 + lane];
    v16h kpn1 = kpB[(nk * 2 + 1) * 32 + lane];
    v16h vpn0 = vpB[nk * 64 + lane];
    v16h vpn1 = vpB[nk * 64 + 32 + lane];

    v8f z = {};
    v8f s0 = wmma16(qa, kpc0, z);
    v8f s1 = wmma16(qa, kpc1, z);

    int kpos0 = kb * 32 + col, kpos1 = kpos0 + 16;
    float mb0 = 1e9f * (mask[(size_t)b * NRES + kpos0] - 1.0f);
    float mb1 = 1e9f * (mask[(size_t)b * NRES + kpos1] - 1.0f);
    float sv0[8], sv1[8], sc[8];
#pragma unroll
    for (int v = 0; v < 8; ++v) {
      int q = q0 + v + 8 * hi;
      sv0[v] = s0[v] + mb0 + NB[nbBase + (size_t)q * NRES + kpos0];
      sv1[v] = s1[v] + mb1 + NB[nbBase + (size_t)q * NRES + kpos1];
    }
#pragma unroll
    for (int v = 0; v < 8; ++v) {
      float r = fmaxf(sv0[v], sv1[v]);
#pragma unroll
      for (int off = 8; off >= 1; off >>= 1)   // stay inside 16-lane row group
        r = fmaxf(r, __shfl_xor(r, off, 32));
      float mn = fmaxf(rmax[v], r);
      sc[v] = __expf(rmax[v] - mn);
      rmax[v] = mn;
      float p0 = __expf(sv0[v] - mn);
      float p1 = __expf(sv1[v] - mn);
      float rs = p0 + p1;
#pragma unroll
      for (int off = 8; off >= 1; off >>= 1)
        rs += __shfl_xor(rs, off, 32);
      lsum[v] = lsum[v] * sc[v] + rs;
      Ps[h][(v + 8 * hi) * KDIM + col]      = (f16_t)p0;
      Ps[h][(v + 8 * hi) * KDIM + 16 + col] = (f16_t)p1;
      oc0[v] *= sc[v];
      oc1[v] *= sc[v];
    }
    __asm__ volatile("s_wait_dscnt 0x0" ::: "memory");
    v16h pa = load_a16x32(Ps[h] + (lane & 15) * KDIM + (lane >> 4) * 16);
    oc0 = wmma16(pa, vpc0, oc0);
    oc1 = wmma16(pa, vpc1, oc1);

    kpc0 = kpn0; kpc1 = kpn1; vpc0 = vpn0; vpc1 = vpn1;
  }

  // normalize, gate (in-register: layouts match), stage gated O as A-frag
#pragma unroll
  for (int v = 0; v < 8; ++v) {
    float inv = 1.0f / lsum[v];
    float g0 = 1.0f / (1.0f + __expf(-(gf[0][v] + gate_b[h * KDIM + col])));
    float g1 = 1.0f / (1.0f + __expf(-(gf[1][v] + gate_b[h * KDIM + 16 + col])));
    Ps[h][(v + 8 * hi) * KDIM + col]      = (f16_t)(oc0[v] * inv * g0);
    Ps[h][(v + 8 * hi) * KDIM + 16 + col] = (f16_t)(oc1[v] * inv * g1);
  }
  __asm__ volatile("s_wait_dscnt 0x0" ::: "memory");
  v16h oa = load_a16x32(Ps[h] + (lane & 15) * KDIM + (lane >> 4) * 16);

  // output projection: wave h consumes o_w's K-step h; reduce heads via LDS
#pragma unroll
  for (int nt = 0; nt < 8; ++nt) {
    v8f z = {};
    v16h bf = *(const v16h*)(WPo + (size_t)(((h * 8 + nt) * 32 + lane) * 16));
    v8f part = wmma16(oa, bf, z);
#pragma unroll
    for (int v = 0; v < 8; ++v)
      atomicAdd(&outAcc[(v + 8 * hi) * CDIM + nt * 16 + col], part[v]);
  }
  __syncthreads();
  float* op = out + ((size_t)b * NRES + q0) * CDIM;
  for (int i = tid; i < 16 * CDIM; i += 128) op[i] = outAcc[i];
}

// ---------------------------------------------------------------------------
extern "C" void kernel_launch(void* const* d_in, const int* in_sizes, int n_in,
                              void* d_out, int out_size, void* d_ws, size_t ws_size,
                              hipStream_t stream) {
  (void)in_sizes; (void)n_in; (void)out_size; (void)ws_size;
  const float* pair_act = (const float*)d_in[0];
  const float* pair_mask = (const float*)d_in[1];
  const float* ln_scale = (const float*)d_in[2];
  const float* ln_bias = (const float*)d_in[3];
  const float* feat2d_w = (const float*)d_in[4];
  const float* q_w = (const float*)d_in[5];
  const float* k_w = (const float*)d_in[6];
  const float* v_w = (const float*)d_in[7];
  const float* gate_w = (const float*)d_in[8];
  const float* gate_b = (const float*)d_in[9];
  const float* o_w = (const float*)d_in[10];
  const float* o_b = (const float*)d_in[11];
  float* out = (float*)d_out;

  // workspace carve-up (all 256B aligned by construction)
  const size_t SZ_X16 = (size_t)NN * CDIM * sizeof(f16_t);           // 37.75 MB
  const size_t SZ_NB  = (size_t)NHEAD * NN * sizeof(float);          //  2.36 MB
  const size_t SZ_KP  = (size_t)NN * NHEAD * KDIM * sizeof(f16_t);   // 37.75 MB
  const size_t SZ_VP  = SZ_KP;                                       // 37.75 MB
  char* ws = (char*)d_ws;
  f16_t* X16 = (f16_t*)ws;                 ws += SZ_X16;
  float* NB  = (float*)ws;                 ws += SZ_NB;
  f16_t* KP  = (f16_t*)ws;                 ws += SZ_KP;
  f16_t* VP  = (f16_t*)ws;                 ws += SZ_VP;
  f16_t* WP  = (f16_t*)ws;                 // 5 * 128*128 f16 = 160 KB
  f16_t* WPq = WP;
  f16_t* WPk = WP + 1 * CDIM * CDIM;
  f16_t* WPv = WP + 2 * CDIM * CDIM;
  f16_t* WPg = WP + 3 * CDIM * CDIM;
  f16_t* WPo = WP + 4 * CDIM * CDIM;

  pack_w_kernel<<<dim3(5), 1024, 0, stream>>>(q_w, k_w, v_w, gate_w, o_w, WP);
  ln_kernel<<<dim3(NN / 8), 256, 0, stream>>>(pair_act, ln_scale, ln_bias,
                                              feat2d_w, X16, NB);
  const int kvLds = 2 * NRES * CDIM * (int)sizeof(f16_t);  // 192 KB of 320 KB WGP LDS
  hipFuncSetAttribute((const void*)kv_proj_kernel,
                      hipFuncAttributeMaxDynamicSharedMemorySize, kvLds);
  kv_proj_kernel<<<dim3(NRES), 256, kvLds, stream>>>(X16, WPk, WPv, KP, VP);
  att_kernel<<<dim3(NRES / 16, NRES), 128, 0, stream>>>(
      X16, NB, pair_mask, KP, VP, WPq, WPg, WPo, gate_b, o_b, out);
}
